// ImplicitFeatureAlignment_17815524343851
// MI455X (gfx1250) — matrix operand order repaired and checked by reading, not compiled, not run on hardware
//
#include <hip/hip_runtime.h>
#include <hip/hip_bf16.h>
#include <cstdint>

typedef __attribute__((ext_vector_type(16))) __bf16 v16bf;
typedef __attribute__((ext_vector_type(8)))  __bf16 v8bf;
typedef __attribute__((ext_vector_type(8)))  float  v8f;

#define NPIX   65536   // 4 * 128 * 128
#define C_CTX  1064
#define K1     1088    // 1064 padded up to 34*32
#define F1     512
#define K2     512
#define F2     256
#define K3     256
#define F3     256
#define BN_EPS 1e-3f

__device__ __forceinline__ __bf16 f2bf(float f) {
  union { float f; uint32_t u; } in; in.f = f;
  uint32_t r = in.u + 0x7FFFu + ((in.u >> 16) & 1u);   // round-to-nearest-even
  union { uint16_t s; __bf16 b; } out; out.s = (uint16_t)(r >> 16);
  return out.b;
}

// Low 32 bits of a flat shared-memory address are the LDS byte address
// (ISA 10.2 aperture mapping).
__device__ __forceinline__ uint32_t lds_off(const void* p) {
  return (uint32_t)(uintptr_t)p;
}

// CDNA5 async DMA: copy 16 bytes/lane global -> LDS, tracked by ASYNCcnt.
__device__ __forceinline__ void async_copy_b128(uint32_t lds_addr, const void* gptr) {
  asm volatile("global_load_async_to_lds_b128 %0, %1, off"
               :: "v"(lds_addr), "v"((unsigned long long)(uintptr_t)gptr)
               : "memory");
}

// ---------------------------------------------------------------------------
// Build context: nearest-sample 4 pyramid levels + positional encoding -> bf16
// ---------------------------------------------------------------------------
__global__ __launch_bounds__(256) void build_context_kernel(
    const float* __restrict__ f0, const float* __restrict__ f1,
    const float* __restrict__ f2, const float* __restrict__ f3,
    const float* __restrict__ emb, __bf16* __restrict__ ctx)
{
  const int n   = blockIdx.x;
  const int tid = threadIdx.x;
  const int b = n >> 14;
  const int y = (n >> 7) & 127;
  const int x = n & 127;
  const float cy = (y + 0.5f) * (2.0f / 128.0f) - 1.0f;
  const float cx = (x + 0.5f) * (2.0f / 128.0f) - 1.0f;

  const int   Hs[4]  = {128, 64, 32, 16};
  const int   Cs[4]  = {64, 128, 256, 512};
  const int   off[4] = {0, 90, 244, 526};
  const float* feats[4] = {f0, f1, f2, f3};

  __bf16* row = ctx + (size_t)n * K1;

  #pragma unroll
  for (int l = 0; l < 4; ++l) {
    const int h = Hs[l];
    float fy = (cy + 1.0f) * h * 0.5f - 0.5f;
    float fx = (cx + 1.0f) * h * 0.5f - 0.5f;
    int iy = __float2int_rn(fy); iy = iy < 0 ? 0 : (iy > h - 1 ? h - 1 : iy);
    int ix = __float2int_rn(fx); ix = ix < 0 ? 0 : (ix > h - 1 ? h - 1 : ix);
    const float qy = (iy + 0.5f) * (2.0f / h) - 1.0f;
    const float qx = (ix + 0.5f) * (2.0f / h) - 1.0f;
    const float ry = (cy - qy) * h;
    const float rx = (cx - qx) * h;

    const int C = Cs[l];
    const float* src = feats[l] + (((size_t)b * h + iy) * h + ix) * C;
    __bf16* dst = row + off[l];
    for (int c = tid; c < C; c += 256) dst[c] = f2bf(src[c]);

    if (tid < 26) {                           // [rel_y, rel_x, sin(e)x12, cos(e)x12]
      float v;
      if (tid == 0)      v = ry;
      else if (tid == 1) v = rx;
      else if (tid < 14) { int j = tid - 2;  v = sinf(ry * emb[j] + rx * emb[12 + j]); }
      else               { int j = tid - 14; v = cosf(ry * emb[j] + rx * emb[12 + j]); }
      dst[C + tid] = f2bf(v);
    }
  }
  if (tid < K1 - C_CTX) row[C_CTX + tid] = f2bf(0.0f);
}

// ---------------------------------------------------------------------------
// Weights f32 [Ci][Co] -> bf16 transposed [Co][Kpad] (zero padded K rows)
// ---------------------------------------------------------------------------
__global__ void convert_w_kernel(const float* __restrict__ W, __bf16* __restrict__ Wt,
                                 int Ci, int Co, int Kpad)
{
  int idx = blockIdx.x * 256 + threadIdx.x;
  if (idx >= Co * Kpad) return;
  int f = idx / Kpad, k = idx - f * Kpad;
  Wt[idx] = (k < Ci) ? f2bf(W[(size_t)k * Co + f]) : f2bf(0.0f);
}

__global__ void zero_f32_kernel(float* p, int n) {
  int i = blockIdx.x * 256 + threadIdx.x;
  if (i < n) p[i] = 0.0f;
}

// ---------------------------------------------------------------------------
// GEMM  Y[N x F] = A[N x K] * Wt[F x K]^T  via v_wmma_f32_16x16x32_bf16.
// A/B tiles staged with GLOBAL_LOAD_ASYNC_TO_LDS_B128 (ASYNCcnt), double
// buffered; the last iteration is peeled so the steady-state loop body is
// branch-free: issue tile k+1, s_wait_asynccnt<=2 (in-order completion =>
// tile k landed), barrier, WMMA, barrier.
// Per-output-channel sum/sumsq accumulated for batch-norm statistics.
// Block: 256 threads = 8 wave32s; block tile 64(M) x 64(F); wave tile 16x32.
// ---------------------------------------------------------------------------
__global__ __launch_bounds__(256) void gemm_async_bn_stats(
    const __bf16* __restrict__ A, const __bf16* __restrict__ Wt,
    float* __restrict__ Y, float* __restrict__ gSum, float* __restrict__ gSq,
    int K, int F)
{
  __shared__ __align__(128) __bf16 As[2][64][32];
  __shared__ __align__(128) __bf16 Bs[2][64][32];   // Bs[buf][col][k]
  __shared__ float sSum[64];
  __shared__ float sSq[64];

  const int tid  = threadIdx.x;
  const int lane = tid & 31;
  const int wave = tid >> 5;
  const int wm   = wave & 3;     // 4 row-tiles of 16
  const int wn   = wave >> 2;    // 2 col-tiles of 32
  const int half = lane >> 4;
  const int nn   = lane & 15;
  const int    col0 = blockIdx.x * 64;          // F tiles fastest -> A reuse in L2
  const size_t row0 = (size_t)blockIdx.y * 64;

  if (tid < 64) { sSum[tid] = 0.0f; sSq[tid] = 0.0f; }

  v8f c0 = {}, c1 = {};
  const int r  = tid >> 2;         // 0..63 : tile row / weight column
  const int c8 = (tid & 3) * 8;    // 8 K-elements (16 bytes) per thread

  const __bf16* gA = A  + (row0 + r) * (size_t)K + c8;
  const __bf16* gB = Wt + (size_t)(col0 + r) * K + c8;

  // One k-step of fragment loads + WMMA math on LDS buffer `buf`.
  auto compute_tile = [&](int buf) {
    v16bf a;
    {
      const v8bf alo = *(const v8bf*)&As[buf][wm * 16 + nn][half * 16];
      const v8bf ahi = *(const v8bf*)&As[buf][wm * 16 + nn][half * 16 + 8];
      #pragma unroll
      for (int i = 0; i < 8; ++i) { a[i] = alo[i]; a[8 + i] = ahi[i]; }
    }
    const v16bf b0 = *(const v16bf*)&Bs[buf][wn * 32 + nn][half * 16];
    const v16bf b1 = *(const v16bf*)&Bs[buf][wn * 32 + 16 + nn][half * 16];
    c0 = __builtin_amdgcn_wmma_f32_16x16x32_bf16(false, a, false, b0, (short)0, c0,
                                                 false, false);
    c1 = __builtin_amdgcn_wmma_f32_16x16x32_bf16(false, a, false, b1, (short)0, c1,
                                                 false, false);
  };

  // prologue: tile 0 -> buffer 0
  async_copy_b128(lds_off(&As[0][r][c8]), gA);
  async_copy_b128(lds_off(&Bs[0][r][c8]), gB);

  const int KT = K >> 5;
  for (int kt = 0; kt < KT - 1; ++kt) {        // steady state: branch-free body
    const int buf = kt & 1;
    async_copy_b128(lds_off(&As[buf ^ 1][r][c8]), gA + (size_t)(kt + 1) * 32);
    async_copy_b128(lds_off(&Bs[buf ^ 1][r][c8]), gB + (size_t)(kt + 1) * 32);
    asm volatile("s_wait_asynccnt 0x2" ::: "memory");   // tile kt landed (in-order)
    __syncthreads();
    compute_tile(buf);
    __syncthreads();                  // protect buf before it is overwritten
  }
  // peeled last iteration: nothing left to prefetch
  asm volatile("s_wait_asynccnt 0x0" ::: "memory");
  __syncthreads();
  compute_tile((KT - 1) & 1);

  // Epilogue: store f32 result + per-channel stats (LDS partial, global atomics)
  const size_t mbase = row0 + wm * 16 + half * 8;   // D VGPR i -> M = i + half*8
  float s0 = 0, q0 = 0, s1 = 0, q1 = 0;
  #pragma unroll
  for (int i = 0; i < 8; ++i) {
    const float v0 = c0[i], v1 = c1[i];
    Y[(mbase + i) * F + col0 + wn * 32 + nn]      = v0;
    Y[(mbase + i) * F + col0 + wn * 32 + 16 + nn] = v1;
    s0 += v0; q0 += v0 * v0; s1 += v1; q1 += v1 * v1;
  }
  atomicAdd(&sSum[wn * 32 + nn], s0);      atomicAdd(&sSq[wn * 32 + nn], q0);
  atomicAdd(&sSum[wn * 32 + 16 + nn], s1); atomicAdd(&sSq[wn * 32 + 16 + nn], q1);
  __syncthreads();
  if (tid < 64) {
    atomicAdd(&gSum[col0 + tid], sSum[tid]);
    atomicAdd(&gSq[col0 + tid],  sSq[tid]);
  }
}

__global__ void finalize_bn_kernel(const float* __restrict__ sum, const float* __restrict__ sq,
                                   const float* __restrict__ g,   const float* __restrict__ bb,
                                   float* __restrict__ scale, float* __restrict__ shift,
                                   int F, float invN)
{
  int ch = blockIdx.x * 256 + threadIdx.x;
  if (ch >= F) return;
  float mu  = sum[ch] * invN;
  float var = sq[ch] * invN - mu * mu;
  float a   = g[ch] * rsqrtf(var + BN_EPS);
  scale[ch] = a;
  shift[ch] = bb[ch] - a * mu;
}

// Apply BN + ReLU, emit bf16 activations for the next GEMM's async path.
__global__ void bn_relu_bf16_kernel(const float* __restrict__ y,
                                    const float* __restrict__ scale,
                                    const float* __restrict__ shift,
                                    __bf16* __restrict__ x, int total, int chMask)
{
  int i = blockIdx.x * 256 + threadIdx.x;
  if (i >= total) return;
  int ch = i & chMask;
  x[i] = f2bf(fmaxf(0.0f, scale[ch] * y[i] + shift[ch]));
}

__global__ void bn_relu_out_kernel(const float* __restrict__ y,
                                   const float* __restrict__ scale,
                                   const float* __restrict__ shift,
                                   float* __restrict__ out, int total)
{
  int i = blockIdx.x * 256 + threadIdx.x;
  if (i >= total) return;
  int ch = i & (F3 - 1);
  out[i] = fmaxf(0.0f, scale[ch] * y[i] + shift[ch]);
}

// ---------------------------------------------------------------------------
extern "C" void kernel_launch(void* const* d_in, const int* in_sizes, int n_in,
                              void* d_out, int out_size, void* d_ws, size_t ws_size,
                              hipStream_t stream) {
  const float* feat0 = (const float*)d_in[0];
  const float* feat1 = (const float*)d_in[1];
  const float* feat2 = (const float*)d_in[2];
  const float* feat3 = (const float*)d_in[3];
  const float* emb   = (const float*)d_in[4];
  const float* w1 = (const float*)d_in[8];
  const float* g1 = (const float*)d_in[9];
  const float* b1 = (const float*)d_in[10];
  const float* w2 = (const float*)d_in[11];
  const float* g2 = (const float*)d_in[12];
  const float* b2 = (const float*)d_in[13];
  const float* w3 = (const float*)d_in[14];
  const float* g3 = (const float*)d_in[15];
  const float* b3 = (const float*)d_in[16];

  char* ws = (char*)d_ws;
  size_t off = 0;
  auto take = [&](size_t bytes) -> void* {
    void* p = ws + off;
    off = (off + bytes + 255) & ~(size_t)255;
    return p;
  };
  __bf16* ctx = (__bf16*)take((size_t)NPIX * K1 * 2);   // 142.6 MB
  __bf16* w1t = (__bf16*)take((size_t)F1 * K1 * 2);
  __bf16* w2t = (__bf16*)take((size_t)F2 * K2 * 2);
  __bf16* w3t = (__bf16*)take((size_t)F3 * K3 * 2);
  float*  y1  = (float*)take((size_t)NPIX * F1 * 4);    // 134 MB
  float*  y2  = (float*)take((size_t)NPIX * F2 * 4);    // 67 MB
  float*  stats = (float*)take(2048 * 4);
  float *sum1 = stats,        *sq1 = stats + 512;
  float *sum2 = stats + 1024, *sq2 = stats + 1280;
  float *sum3 = stats + 1536, *sq3 = stats + 1792;
  float*  ss = (float*)take(2048 * 4);
  float *scale1 = ss,        *shift1 = ss + 512;
  float *scale2 = ss + 1024, *shift2 = ss + 1280;
  float *scale3 = ss + 1536, *shift3 = ss + 1792;

  // Aliases into dead buffers (stream-ordered, no overlap at time of use):
  __bf16* x1 = (__bf16*)ctx;                                  // ctx dead after GEMM1
  __bf16* x2 = (__bf16*)((char*)ctx + (size_t)NPIX * F1 * 2 + 4096);
  float*  y3 = y1;                                            // y1 dead after x1 built
  float*  out = (float*)d_out;
  const float invN = 1.0f / (float)NPIX;

  zero_f32_kernel<<<8, 256, 0, stream>>>(stats, 2048);
  convert_w_kernel<<<(F1 * K1 + 255) / 256, 256, 0, stream>>>(w1, w1t, C_CTX, F1, K1);
  convert_w_kernel<<<(F2 * K2 + 255) / 256, 256, 0, stream>>>(w2, w2t, K2, F2, K2);
  convert_w_kernel<<<(F3 * K3 + 255) / 256, 256, 0, stream>>>(w3, w3t, K3, F3, K3);
  build_context_kernel<<<NPIX, 256, 0, stream>>>(feat0, feat1, feat2, feat3, emb, ctx);

  dim3 grid1(F1 / 64, NPIX / 64);
  gemm_async_bn_stats<<<grid1, 256, 0, stream>>>(ctx, w1t, y1, sum1, sq1, K1, F1);
  finalize_bn_kernel<<<2, 256, 0, stream>>>(sum1, sq1, g1, b1, scale1, shift1, F1, invN);
  bn_relu_bf16_kernel<<<(NPIX * F1 + 255) / 256, 256, 0, stream>>>(
      y1, scale1, shift1, x1, NPIX * F1, F1 - 1);

  dim3 grid2(F2 / 64, NPIX / 64);
  gemm_async_bn_stats<<<grid2, 256, 0, stream>>>(x1, w2t, y2, sum2, sq2, K2, F2);
  finalize_bn_kernel<<<1, 256, 0, stream>>>(sum2, sq2, g2, b2, scale2, shift2, F2, invN);
  bn_relu_bf16_kernel<<<(NPIX * F2 + 255) / 256, 256, 0, stream>>>(
      y2, scale2, shift2, x2, NPIX * F2, F2 - 1);

  gemm_async_bn_stats<<<grid2, 256, 0, stream>>>(x2, w3t, y3, sum3, sq3, K3, F3);
  finalize_bn_kernel<<<1, 256, 0, stream>>>(sum3, sq3, g3, b3, scale3, shift3, F3, invN);

  bn_relu_out_kernel<<<(NPIX * F3 + 255) / 256, 256, 0, stream>>>(y3, scale3, shift3,
                                                                  out, NPIX * F3);
}
